// lfc_87522843558243
// MI455X (gfx1250) — 8-bit matrix operand order repaired and checked by reading, compiled, not run on hardware
//
#include <hip/hip_runtime.h>
#include <stdint.h>


typedef __attribute__((ext_vector_type(16))) int   v16i;
typedef __attribute__((ext_vector_type(8)))  float v8f;

#define BN_EPS 1e-5f

__device__ __forceinline__ uint8_t sgn_fp8(float v) {
    return v > 0.f ? 0x38 : (v < 0.f ? 0xB8 : 0x00);
}

__global__ void binarize_pad(const float* __restrict__ in, uint8_t* __restrict__ out,
                             int rows, int cols, int rows_pad, int cols_pad) {
    size_t idx = (size_t)blockIdx.x * blockDim.x + threadIdx.x;
    size_t total = (size_t)rows_pad * cols_pad;
    if (idx >= total) return;
    int c = (int)(idx % cols_pad);
    int r = (int)(idx / cols_pad);
    uint8_t v = 0;
    if (r < rows && c < cols) v = sgn_fp8(in[(size_t)r * cols + c]);
    out[idx] = v;
}

template<int MT, int NT, int WN, bool FINAL>
__global__ __launch_bounds__(256)
void bnn_gemm(const uint8_t* __restrict__ A, const uint8_t* __restrict__ W,
              const float* __restrict__ bias, const float* __restrict__ gamma,
              const float* __restrict__ beta, const float* __restrict__ mean,
              const float* __restrict__ var,
              uint8_t* __restrict__ outq, float* __restrict__ outf,
              int Kpad, int out_ld, int n_valid)
{
    constexpr int WM = 8 / WN;
    const int lane  = threadIdx.x & 31;
    const int wave  = threadIdx.x >> 5;
    const int wm    = wave % WM;
    const int wn    = wave / WM;
    const int m0    = blockIdx.y * (16 * MT * WM) + wm * (16 * MT);
    const int n0    = blockIdx.x * (16 * NT * WN) + wn * (16 * NT);
    const int half  = (lane & 16) ? 1 : 0;
    const int nlane = lane & 15;

    const uint8_t* aptr[MT];
#pragma unroll
    for (int mt = 0; mt < MT; ++mt)
        aptr[mt] = A + (size_t)(m0 + mt * 16 + nlane) * Kpad + (half ? 16 : 0);

    v8f acc[MT][NT];
#pragma unroll
    for (int mt = 0; mt < MT; ++mt)
#pragma unroll
        for (int t = 0; t < NT; ++t) acc[mt][t] = (v8f)0.f;

    for (int k0 = 0; k0 < Kpad; k0 += 128) {
        v16i av[MT];
#pragma unroll
        for (int mt = 0; mt < MT; ++mt) {
            const uint8_t* ak = aptr[mt] + k0;
            int4 a0 = *(const int4*)(ak);        int4 a1 = *(const int4*)(ak + 32);
            int4 a2 = *(const int4*)(ak + 64);   int4 a3 = *(const int4*)(ak + 96);
            av[mt] = (v16i){ a0.x, a0.y, a0.z, a0.w, a1.x, a1.y, a1.z, a1.w,
                             a2.x, a2.y, a2.z, a2.w, a3.x, a3.y, a3.z, a3.w };
        }
#pragma unroll
        for (int t = 0; t < NT; ++t) {
            int n = n0 + t * 16 + nlane;
            const uint8_t* wk = W + (size_t)n * Kpad + k0 + (half ? 16 : 0);
            int4 b0 = *(const int4*)(wk);
            int4 b1 = *(const int4*)(wk + 32);
            int4 b2 = *(const int4*)(wk + 64);
            int4 b3 = *(const int4*)(wk + 96);
            v16i bv = { b0.x, b0.y, b0.z, b0.w, b1.x, b1.y, b1.z, b1.w,
                        b2.x, b2.y, b2.z, b2.w, b3.x, b3.y, b3.z, b3.w };
#pragma unroll
            for (int mt = 0; mt < MT; ++mt)
                acc[mt][t] = __builtin_amdgcn_wmma_f32_16x16x128_fp8_fp8(
                    av[mt], bv, (short)0, acc[mt][t], false, false);
        }
    }

#pragma unroll
    for (int t = 0; t < NT; ++t) {
        int n  = n0 + t * 16 + nlane;
        int np = n < n_valid ? n : n_valid - 1;
        float bs = bias[np];
        float sc = gamma[np] * rsqrtf(var[np] + BN_EPS);
        float sh = beta[np] - sc * mean[np];
#pragma unroll
        for (int mt = 0; mt < MT; ++mt) {
#pragma unroll
            for (int r = 0; r < 8; ++r) {
                int m   = m0 + mt * 16 + r + half * 8;
                float y = sc * (acc[mt][t][r] + bs) + sh;
                if (FINAL) {
                    if (n < n_valid) outf[(size_t)m * n_valid + n] = y;
                } else {
                    outq[(size_t)m * out_ld + n] = sgn_fp8(y);
                }
            }
        }
    }
}

extern "C" void kernel_launch(void* const* d_in, const int* in_sizes, int n_in,
                              void* d_out, int out_size, void* d_ws, size_t ws_size,
                              hipStream_t stream) {
    (void)in_sizes; (void)n_in; (void)out_size; (void)ws_size;
    const float* x = (const float*)d_in[0];
    const float *W[4], *b[4], *g[4], *be[4], *mn[4], *vr[4];
    for (int i = 0; i < 4; ++i) {
        int base = 1 + 6 * i;
        W[i]  = (const float*)d_in[base + 0];
        b[i]  = (const float*)d_in[base + 1];
        g[i]  = (const float*)d_in[base + 2];
        be[i] = (const float*)d_in[base + 3];
        mn[i] = (const float*)d_in[base + 4];
        vr[i] = (const float*)d_in[base + 5];
    }

    const int M = 32768;
    uint8_t* ws   = (uint8_t*)d_ws;
    uint8_t* bufA = ws;
    uint8_t* bufB = ws + 33554432ull;
    uint8_t* W1s  = ws + 67108864ull;
    uint8_t* W2s  = W1s + 1024ull * 896;
    uint8_t* W3s  = W2s + 1024ull * 1024;
    uint8_t* W4s  = W3s + 1024ull * 1024;

    auto bgrid = [](size_t total) { return (unsigned)((total + 255) / 256); };

    binarize_pad<<<bgrid((size_t)M * 896), 256, 0, stream>>>(x,    bufA, M,    784, M,    896);
    binarize_pad<<<bgrid(1024ull * 896),  256, 0, stream>>>(W[0], W1s, 1024, 784, 1024, 896);
    binarize_pad<<<bgrid(1024ull * 1024), 256, 0, stream>>>(W[1], W2s, 1024, 1024, 1024, 1024);
    binarize_pad<<<bgrid(1024ull * 1024), 256, 0, stream>>>(W[2], W3s, 1024, 1024, 1024, 1024);
    binarize_pad<<<bgrid(16ull * 1024),   256, 0, stream>>>(W[3], W4s, 10,   1024, 16,   1024);

    dim3 blk(256);
    bnn_gemm<2, 4, 2, false><<<dim3(8, 256), blk, 0, stream>>>(
        bufA, W1s, b[0], g[0], be[0], mn[0], vr[0], bufB, nullptr, 896, 1024, 1024);
    bnn_gemm<2, 4, 2, false><<<dim3(8, 256), blk, 0, stream>>>(
        bufB, W2s, b[1], g[1], be[1], mn[1], vr[1], bufA, nullptr, 1024, 1024, 1024);
    bnn_gemm<2, 4, 2, false><<<dim3(8, 256), blk, 0, stream>>>(
        bufA, W3s, b[2], g[2], be[2], mn[2], vr[2], bufB, nullptr, 1024, 1024, 1024);
    bnn_gemm<2, 1, 1, true><<<dim3(1, 128), blk, 0, stream>>>(
        bufB, W4s, b[3], g[3], be[3], mn[3], vr[3], nullptr, (float*)d_out, 1024, 16, 10);
}
